// GaussianKernel_70437463654936
// MI455X (gfx1250) — matrix operand order repaired and checked by reading, not compiled, not run on hardware
//
#include <hip/hip_runtime.h>
#include <hip/hip_bf16.h>

typedef __attribute__((ext_vector_type(16))) _Float16 v16h;
typedef __attribute__((ext_vector_type(8)))  float    v8f;

#define BH     96            // B*H = 8*12
#define NTOK   577
#define HDIM   64
#define TOKENS (BH * NTOK)   // 55392
#define TILES  (TOKENS / 16) // 3462 (exact)

__global__ __launch_bounds__(256)
void gaussian_mask_fused_70437463654936(const float* __restrict__ q,
                                        const float* __restrict__ W1,
                                        const float* __restrict__ b1,
                                        const float* __restrict__ W2,
                                        const float* __restrict__ b2,
                                        const float* __restrict__ u,
                                        const float* __restrict__ dists,
                                        float* __restrict__ out)
{
    __shared__ float hbuf[16 * 64];            // gelu(q@W1+b1) for 16 tokens
    __shared__ float ca[16], cm[16], cb[16];   // folded quadratic-form coeffs

    const int tile    = blockIdx.x;
    const int tokBase = tile * 16;
    const int lane    = threadIdx.x & 31;
    const int wave    = threadIdx.x >> 5;

    // ---------------- Phase 1: 16-token MLP via WMMA (wave 0 only) ----------
    if (wave == 0) {
        const int rowM     = lane & 15;
        const int kbA      = (lane >> 4) << 4;   // 0 or 16  (A K-split)
        const int kbB      = (lane >> 4) << 4;   // 0 or 16  (B K-split)
        const int ncolBase = lane & 15;

        v8f acc[4] = {};
#pragma unroll
        for (int ks = 0; ks < 2; ++ks) {
            const int k0 = ks * 32;
            // A fragment: 16x32 f16, K contiguous kbA..kbA+15 (matches B)
            const float* qrow = q + (size_t)(tokBase + rowM) * HDIM + k0;
            v16h a;
#pragma unroll
            for (int h = 0; h < 16; ++h) {
                const int K = kbA + h;
                a[h] = (_Float16)qrow[K];
            }
#pragma unroll
            for (int nt = 0; nt < 4; ++nt) {
                const int ncol = nt * 16 + ncolBase;
                // B fragment: 32x16 f16, column ncol, K rows kbB..kbB+15
                v16h bf;
#pragma unroll
                for (int h = 0; h < 16; ++h)
                    bf[h] = (_Float16)W1[(size_t)(k0 + kbB + h) * 64 + ncol];
                acc[nt] = __builtin_amdgcn_wmma_f32_16x16x32_f16(
                    false, a, false, bf, (short)0, acc[nt], false, false);
            }
        }
        // epilogue: +b1, exact gelu, scatter C layout (M = v + 8*lane[4]) to LDS
        const int mBase = (lane >> 4) << 3;
#pragma unroll
        for (int nt = 0; nt < 4; ++nt) {
            const int   ncol = nt * 16 + ncolBase;
            const float bias = b1[ncol];
#pragma unroll
            for (int v = 0; v < 8; ++v) {
                float x = acc[nt][v] + bias;
                float g = 0.5f * x * (1.0f + erff(x * 0.70710678118654752f));
                hbuf[(mBase + v) * 64 + ncol] = g;
            }
        }
    }
    __syncthreads();

    // ------------- Phase 1b: W2/b2 + closed-form 2x2 inverse (16 lanes) -----
    if (threadIdx.x < 16) {
        float s0 = b2[0], s1 = b2[1], s2 = b2[2];
        const float* hr = hbuf + threadIdx.x * 64;
#pragma unroll 8
        for (int k = 0; k < 64; ++k) {
            const float hv = hr[k];
            s0 = fmaf(hv, W2[k * 3 + 0], s0);
            s1 = fmaf(hv, W2[k * 3 + 1], s1);
            s2 = fmaf(hv, W2[k * 3 + 2], s2);
        }
        const float sx  = fmaxf(s0, 0.0f) + 1.0f;
        const float sy  = fmaxf(s1, 0.0f) + 1.0f;
        const float rho = 0.99f * tanhf(s2);
        const float sxx = sx * sx, syy = sy * sy, sxy = rho * sx * sy;
        const float det = sxx * syy - sxy * sxy;
        const float rdt = 1.0f / det;
        // fold -0.5 (and the 2*inv01 sign) into the coefficients:
        // w = ca*dx^2 + cm*dx*dy + cb*dy^2
        ca[threadIdx.x] = -0.5f * syy * rdt;
        cb[threadIdx.x] = -0.5f * sxx * rdt;
        cm[threadIdx.x] =  sxy * rdt;         // = -inv01
    }
    __syncthreads();

    // ------------- Phase 2: stream 16 rows x 577 cols (all 8 waves) ---------
    // rowmax(kernel) == 1 analytically (w<=0, w==0 at j==0)  -> no reduction.
    // sigmoid((logit+noise)/0.1) == 1/(1 + r^10),
    //   r = ((1-p)/p)*((1-u)/u),  p = exp(clamp(w, ln eps, ln(1-eps)))
    const float2* __restrict__ dp = (const float2*)dists;
#pragma unroll 1
    for (int r = 0; r < 16; ++r) {
        const int   t  = tokBase + r;
        const int   i  = t % NTOK;
        const float a2 = ca[r];
        const float m2 = cm[r];
        const float b2v = cb[r];
        const float2* __restrict__ drow = dp + (size_t)i * NTOK;
        const float*  __restrict__ urow = u  + (size_t)t * NTOK;
        float*        __restrict__ orow = out + (size_t)t * NTOK;
        for (int j = (int)threadIdx.x; j < NTOK; j += 256) {
            const float2 d  = drow[j];
            const float  dx = d.x, dy = d.y;
            float w = a2 * (dx * dx);
            w = fmaf(m2, dx * dy, w);
            w = fmaf(b2v, dy * dy, w);
            const float wc = fminf(fmaxf(w, -13.815510557964274f), -1.0000005e-6f);
            const float p  = __expf(wc);                 // v_exp_f32
            const float uu = urow[j];
            const float num = (1.0f - p) * (1.0f - uu);  // (1-p)(1-u)
            const float den = p * uu;                    // p*u
            const float rr  = num * __builtin_amdgcn_rcpf(den);
            const float r2  = rr * rr;
            const float r4  = r2 * r2;
            const float r8  = r4 * r4;
            const float r10 = r8 * r2;
            orow[j] = __builtin_amdgcn_rcpf(1.0f + r10); // sigmoid(z/0.1)
        }
    }
}

extern "C" void kernel_launch(void* const* d_in, const int* in_sizes, int n_in,
                              void* d_out, int out_size, void* d_ws, size_t ws_size,
                              hipStream_t stream) {
    (void)in_sizes; (void)n_in; (void)d_ws; (void)ws_size; (void)out_size;
    const float* q  = (const float*)d_in[0];
    const float* W1 = (const float*)d_in[1];
    const float* b1 = (const float*)d_in[2];
    const float* W2 = (const float*)d_in[3];
    const float* b2 = (const float*)d_in[4];
    const float* u  = (const float*)d_in[5];
    const float* dd = (const float*)d_in[6];
    float* out = (float*)d_out;
    gaussian_mask_fused_70437463654936<<<TILES, 256, 0, stream>>>(
        q, W1, b1, W2, b2, u, dd, out);
}
